// Model_32212254720220
// MI455X (gfx1250) — matrix operand order repaired and checked by reading, not compiled, not run on hardware
//
#include <hip/hip_runtime.h>
#include <hip/hip_bf16.h>

typedef __attribute__((ext_vector_type(16))) _Float16 v16h;
typedef __attribute__((ext_vector_type(8)))  float    v8f;

#define TPB   256
#define ITEMS 32                    // elements per thread
#define TILE  (TPB * ITEMS)         // 8192 elements per block
#define PAGE  16

union HVec {
    v16h     v;
    _Float16 h[16];
    int4     q[2];
};

// ---------------------------------------------------------------------------
// Block-wide scan of 256 per-thread counts (each in [0,32]) using two
// v_wmma_f32_16x16x32_f16 ops on wave 0.
//   T[r][c] = cnt[16r+c]  (16x16)
//   D1 = T x U        (U[k][c] = k<=c)      -> row-wise inclusive scans
//   D2 = L x RT + D1  (L[r][k] = k<r, RT[k][c] = rowtotal(k)) -> full scan
// Returns this thread's EXCLUSIVE prefix; *total = block sum.
// All values stay exactly representable (counts<=32 f16, rowtotals<=512 f16,
// block totals<=8192 f32).
// ---------------------------------------------------------------------------
__device__ __forceinline__ int block_scan_wmma(int cnt, int* total)
{
    __shared__ __align__(16) _Float16 sh_cnt[TPB];
    __shared__ __align__(16) _Float16 sh_rt[16];
    __shared__ __align__(16) float    sh_incl[TPB];

    const int tid = threadIdx.x;
    sh_cnt[tid] = (_Float16)cnt;
    __syncthreads();

    if (tid < 32) {
        const int  l  = tid;
        const bool lo = (l < 16);

        // A[r][k] = cnt[16r+k] for k<16, else 0.
        // lane<16: halves 0..7 = K 0..7 of row l ; lane>=16: K 8..15 of row l-16.
        HVec A;
        A.q[0] = ((const int4*)sh_cnt)[lo ? (2 * l) : (2 * (l - 16) + 1)];
        A.q[1] = make_int4(0, 0, 0, 0);

        // B1[k][c] = (k <= c) for k<16, else 0.  column c=l%16; halves 0..7 = row k=h (lane<16) / k=8+h (lane>=16).
        HVec B1;
        #pragma unroll
        for (int h = 0; h < 16; ++h)
            B1.h[h] = (_Float16)(((h < 8) && ((lo ? h : (8 + h)) <= (l & 15))) ? 1.0f : 0.0f);

        v8f zero = {};
        v8f D1 = __builtin_amdgcn_wmma_f32_16x16x32_f16(
                     false, A.v, false, B1.v, (short)0, zero, false, false);
        // D1[r][c] = sum_{k<=c} cnt[16r+k]

        // row totals rt[r] = D1[r][15]; held by lane 15 (r=0..7) / lane 31 (r=8..15)
        if (l == 15) {
            #pragma unroll
            for (int r = 0; r < 8; ++r) sh_rt[r] = (_Float16)D1[r];
        }
        if (l == 31) {
            #pragma unroll
            for (int r = 0; r < 8; ++r) sh_rt[8 + r] = (_Float16)D1[r];
        }

        // B2[k][c] = rt[k] for k<16 (replicated across columns), else 0
        HVec B2;
        if (lo) {
            B2.q[0] = ((const int4*)sh_rt)[0];
            B2.q[1] = make_int4(0, 0, 0, 0);
        } else {
            B2.q[0] = ((const int4*)sh_rt)[1];
            B2.q[1] = make_int4(0, 0, 0, 0);
        }

        // A2[r][k] = (k < r) for k<16, else 0 (strict lower triangular)
        HVec A2;
        #pragma unroll
        for (int h = 0; h < 16; ++h) {
            int k = (h < 8) ? (lo ? h : (8 + h)) : 99;  // halves 8..15 map to K>=16 -> 0
            int r = lo ? l : (l - 16);
            A2.h[h] = (_Float16)((k < r) ? 1.0f : 0.0f);
        }

        v8f D2 = __builtin_amdgcn_wmma_f32_16x16x32_f16(
                     false, A2.v, false, B2.v, (short)0, D1, false, false);
        // D2[r][c] = inclusive scan of cnt at flat index 16r+c

        #pragma unroll
        for (int r = 0; r < 8; ++r) {
            int row = lo ? r : (r + 8);
            int col = l & 15;
            sh_incl[row * 16 + col] = D2[r];
        }
    }
    __syncthreads();
    *total = (int)sh_incl[TPB - 1];
    return (int)sh_incl[tid] - cnt;
}

// ---------------------------------------------------------------------------
// Pass 1: stream seq_lens once (b128), derive new_page bits, per-block sum.
// Optionally persists the 1-bit-per-element mask so pass 3 never re-reads
// the 64MB seq_lens array (2MB instead).
// ---------------------------------------------------------------------------
__global__ void __launch_bounds__(TPB)
pass1_kernel(const int* __restrict__ seq, unsigned* __restrict__ mask_out,
             int* __restrict__ partials, int n, int write_mask)
{
    const int tid  = threadIdx.x;
    const int base = blockIdx.x * TILE + tid * ITEMS;

    unsigned m = 0u;
    #pragma unroll
    for (int c = 0; c < ITEMS / 4; ++c) {
        int idx = base + c * 4;
        int4 s;
        if (idx + 3 < n) {
            s = *(const int4*)(seq + idx);
        } else {
            s.x = (idx + 0 < n) ? seq[idx + 0] : 0;
            s.y = (idx + 1 < n) ? seq[idx + 1] : 0;
            s.z = (idx + 2 < n) ? seq[idx + 2] : 0;
            s.w = (idx + 3 < n) ? seq[idx + 3] : 0;
        }
        // new_page iff seq_len == 1 (mod 16)
        if ((s.x & 15) == 1) m |= 1u << (c * 4 + 0);
        if ((s.y & 15) == 1) m |= 1u << (c * 4 + 1);
        if ((s.z & 15) == 1) m |= 1u << (c * 4 + 2);
        if ((s.w & 15) == 1) m |= 1u << (c * 4 + 3);
    }

    int total;
    (void)block_scan_wmma(__popc(m), &total);

    if (write_mask) mask_out[blockIdx.x * TPB + tid] = m;
    if (tid == 0)   partials[blockIdx.x] = total;
}

// ---------------------------------------------------------------------------
// Pass 2: exclusive scan of up to 2048 block partials, one 256-thread block.
// Magnitudes reach 2^24 so this stays in int32 (shuffle + LDS scan).
// ---------------------------------------------------------------------------
__global__ void __launch_bounds__(TPB)
pass2_kernel(int* __restrict__ partials, int P)
{
    const int tid  = threadIdx.x;
    const int base = tid * 8;

    int v[8];
    int s = 0;
    #pragma unroll
    for (int j = 0; j < 8; ++j) {
        v[j] = (base + j < P) ? partials[base + j] : 0;
        s += v[j];
    }

    // wave32 inclusive scan of per-thread sums
    const int lane = tid & 31;
    int incl = s;
    #pragma unroll
    for (int d = 1; d < 32; d <<= 1) {
        int t = __shfl_up(incl, d, 32);
        if (lane >= d) incl += t;
    }

    __shared__ int wsum[TPB / 32];
    if (lane == 31) wsum[tid >> 5] = incl;
    __syncthreads();
    if (tid == 0) {
        int acc = 0;
        #pragma unroll
        for (int w = 0; w < TPB / 32; ++w) { int t = wsum[w]; wsum[w] = acc; acc += t; }
    }
    __syncthreads();

    int run = wsum[tid >> 5] + incl - s;   // exclusive prefix of this thread's chunk
    #pragma unroll
    for (int j = 0; j < 8; ++j) {
        int t = v[j];
        if (base + j < P) partials[base + j] = run;
        run += t;
    }
}

// ---------------------------------------------------------------------------
// Pass 3: stream mask (or recompute from seq), last_loc; WMMA block scan gives
// each element its global exclusive page index; gather free_page only where a
// new page is actually needed; write f32 result with b128 stores.
// ---------------------------------------------------------------------------
__global__ void __launch_bounds__(TPB)
pass3_kernel(const unsigned* __restrict__ mask, const int* __restrict__ seq,
             const int* __restrict__ last_loc, const int* __restrict__ free_page,
             const int* __restrict__ partials, float* __restrict__ out,
             int n, int use_mask)
{
    const int tid  = threadIdx.x;
    const int base = blockIdx.x * TILE + tid * ITEMS;

    unsigned m;
    if (use_mask) {
        m = mask[blockIdx.x * TPB + tid];
    } else {
        m = 0u;
        #pragma unroll
        for (int c = 0; c < ITEMS / 4; ++c) {
            int idx = base + c * 4;
            int4 s;
            if (idx + 3 < n) {
                s = *(const int4*)(seq + idx);
            } else {
                s.x = (idx + 0 < n) ? seq[idx + 0] : 0;
                s.y = (idx + 1 < n) ? seq[idx + 1] : 0;
                s.z = (idx + 2 < n) ? seq[idx + 2] : 0;
                s.w = (idx + 3 < n) ? seq[idx + 3] : 0;
            }
            if ((s.x & 15) == 1) m |= 1u << (c * 4 + 0);
            if ((s.y & 15) == 1) m |= 1u << (c * 4 + 1);
            if ((s.z & 15) == 1) m |= 1u << (c * 4 + 2);
            if ((s.w & 15) == 1) m |= 1u << (c * 4 + 3);
        }
    }

    int total;
    const int excl  = block_scan_wmma(__popc(m), &total);
    const int gbase = partials[blockIdx.x] + excl;

    #pragma unroll
    for (int c = 0; c < ITEMS / 4; ++c) {
        int idx = base + c * 4;
        int4 ll;
        if (idx + 3 < n) {
            ll = *(const int4*)(last_loc + idx);
        } else {
            ll.x = (idx + 0 < n) ? last_loc[idx + 0] : 0;
            ll.y = (idx + 1 < n) ? last_loc[idx + 1] : 0;
            ll.z = (idx + 2 < n) ? last_loc[idx + 2] : 0;
            ll.w = (idx + 3 < n) ? last_loc[idx + 3] : 0;
        }
        int lv[4] = {ll.x, ll.y, ll.z, ll.w};
        float ov[4];
        #pragma unroll
        for (int e = 0; e < 4; ++e) {
            int j = c * 4 + e;
            int res;
            if ((m >> j) & 1u) {
                int local = __popc(m & ((1u << j) - 1u));
                res = free_page[gbase + local] * PAGE;   // new page grabbed from free list
            } else {
                res = lv[e] + 1;                          // reuse slot after last_loc
            }
            ov[e] = (float)res;                           // exact: |res| < 2^24
        }
        if (idx + 3 < n) {
            *(float4*)(out + idx) = make_float4(ov[0], ov[1], ov[2], ov[3]);
        } else {
            if (idx + 0 < n) out[idx + 0] = ov[0];
            if (idx + 1 < n) out[idx + 1] = ov[1];
            if (idx + 2 < n) out[idx + 2] = ov[2];
            if (idx + 3 < n) out[idx + 3] = ov[3];
        }
    }
}

// ---------------------------------------------------------------------------
extern "C" void kernel_launch(void* const* d_in, const int* in_sizes, int n_in,
                              void* d_out, int out_size, void* d_ws, size_t ws_size,
                              hipStream_t stream)
{
    const int* seq       = (const int*)d_in[0];
    const int* last_loc  = (const int*)d_in[1];
    const int* free_page = (const int*)d_in[2];
    float*     out       = (float*)d_out;
    const int  n         = in_sizes[0];

    const int nblk = (n + TILE - 1) / TILE;          // 2048 for n = 2^24

    int*      partials = (int*)d_ws;                  // nblk ints at offset 0
    unsigned* maskbuf  = (unsigned*)((char*)d_ws + 16384);
    const size_t need  = 16384 + (size_t)((n + 31) / 32) * sizeof(unsigned);
    const int use_mask = (ws_size >= need) ? 1 : 0;   // constant across calls

    pass1_kernel<<<nblk, TPB, 0, stream>>>(seq, maskbuf, partials, n, use_mask);
    pass2_kernel<<<1, TPB, 0, stream>>>(partials, nblk);
    pass3_kernel<<<nblk, TPB, 0, stream>>>(maskbuf, seq, last_loc, free_page,
                                           partials, out, n, use_mask);
}
